// CharLSTM_7318624272392
// MI455X (gfx1250) — matrix operand order repaired and checked by reading, not compiled, not run on hardware
//
#include <hip/hip_runtime.h>

typedef __attribute__((ext_vector_type(16))) _Float16 v16h;
typedef __attribute__((ext_vector_type(8)))  float    v8f;

#define N_WORDS 16384
#define MAX_LEN 16
#define N_CHARS 128
#define HID     128
#define G4      (4 * HID)   // 512 gate columns

// s_wait_asynccnt <= n : drain this wave's outstanding async-LDS loads
#define WAIT_ASYNCCNT(n) asm volatile("s_wait_asynccnt %0" ::"n"(n) : "memory")

// ---------------------------------------------------------------------------
// Fast activations: gfx1250 has native v_tanh_f32.
//   sigmoid(x) = 0.5*tanh(x/2) + 0.5
// ---------------------------------------------------------------------------
#if __has_builtin(__builtin_amdgcn_tanhf)
__device__ __forceinline__ float fast_tanh(float x) {
    return __builtin_amdgcn_tanhf(x);
}
#else
__device__ __forceinline__ float fast_tanh(float x) {
    // tanh(x) = 1 - 2/(exp2(2x*log2e)+1), native exp2/rcp
    float e = __builtin_amdgcn_exp2f(2.885390082f * x);  // 2*log2(e)
    return 1.0f - 2.0f * __builtin_amdgcn_rcpf(e + 1.0f);
}
#endif
__device__ __forceinline__ float fast_sigmoid(float x) {
    return __builtin_fmaf(0.5f, fast_tanh(0.5f * x), 0.5f);
}

// ---------------------------------------------------------------------------
// Kernel 1: decode one-hot chars -> int indices. One wave32 per (n,t).
// ---------------------------------------------------------------------------
__global__ void decode_chars_kernel(const float* __restrict__ x,
                                    int* __restrict__ idx) {
    const int gwave = (int)((blockIdx.x * blockDim.x + threadIdx.x) >> 5);
    const int lane  = (int)(threadIdx.x & 31);
    if (gwave >= N_WORDS * MAX_LEN) return;
    const float4 v = ((const float4*)(x + (size_t)gwave * N_CHARS))[lane];
    float s = v.x * (float)(lane * 4 + 0) + v.y * (float)(lane * 4 + 1) +
              v.z * (float)(lane * 4 + 2) + v.w * (float)(lane * 4 + 3);
#pragma unroll
    for (int off = 16; off > 0; off >>= 1) s += __shfl_xor(s, off, 32);
    if (lane == 0) idx[gwave] = (int)(s + 0.5f);
}

// ---------------------------------------------------------------------------
// Kernel 2: build lookup/weight tables in workspace.
//   P [dir][c][g]  = W_ih[g*128+c] + b_ih[g] + b_hh[g]   (f32, xproj table)
//   Wt[dir][k][g]  = (f16) W_hh[g*128+k]                 (W_hh^T for WMMA B)
// ---------------------------------------------------------------------------
__global__ void build_tables_kernel(const float* __restrict__ Wih_f,
                                    const float* __restrict__ Whh_f,
                                    const float* __restrict__ bih_f,
                                    const float* __restrict__ bhh_f,
                                    const float* __restrict__ Wih_b,
                                    const float* __restrict__ Whh_b,
                                    const float* __restrict__ bih_b,
                                    const float* __restrict__ bhh_b,
                                    float* __restrict__ P,
                                    _Float16* __restrict__ Wt) {
    const int dir = (int)(blockIdx.x >> 7);
    const int c   = (int)(blockIdx.x & 127);
    const int g   = (int)threadIdx.x;
    const float* Wih = dir ? Wih_b : Wih_f;
    const float* Whh = dir ? Whh_b : Whh_f;
    const float* bih = dir ? bih_b : bih_f;
    const float* bhh = dir ? bhh_b : bhh_f;
    P [((size_t)dir * N_CHARS + c) * G4 + g] = Wih[g * N_CHARS + c] + bih[g] + bhh[g];
    Wt[((size_t)dir * HID     + c) * G4 + g] = (_Float16)Whh[g * HID + c];
}

// ---------------------------------------------------------------------------
// Async gather of one timestep's xproj rows (16 words x 512 f32 = 32KB) from
// the L2-resident P table straight into LDS via GLOBAL_LOAD_ASYNC_TO_LDS_B128
// (ASYNCcnt-tracked DMA, no VGPR round trip). 4 issues per wave.
// ---------------------------------------------------------------------------
__device__ __forceinline__ void async_xproj(const float* __restrict__ Pd,
                                            const int* lidx, int t,
                                            float* buf, int tid) {
#pragma unroll
    for (int pass = 0; pass < 4; ++pass) {
        const int m  = pass * 4 + (tid >> 7);
        const int ci = lidx[m * MAX_LEN + t];
        const unsigned goff = ((unsigned)ci * G4 + (unsigned)(tid & 127) * 4) *
                              (unsigned)sizeof(float);
        const unsigned loff =
            (unsigned)(uintptr_t)(buf + m * G4 + (tid & 127) * 4);
        asm volatile("global_load_async_to_lds_b128 %0, %1, %2"
                     :
                     : "v"(loff), "v"(goff), "s"(Pd)
                     : "memory");
    }
}

// ---------------------------------------------------------------------------
// Kernel 3: fused BiLSTM recurrence (512 threads = 16 waves per 16-word tile).
// Double-buffered gate LDS + async pipelined xproj gather.
// ---------------------------------------------------------------------------
__global__ void __launch_bounds__(512)
bilstm_kernel(const int* __restrict__ idx, const float* __restrict__ P,
              const _Float16* __restrict__ Wt, const int* __restrict__ lengths,
              float* __restrict__ out) {
    __shared__ float    gbuf[2][16 * G4]; // 2x32KB: xproj in / gates out
    __shared__ _Float16 h16[16 * HID];    // 4KB : A operand source
    __shared__ float    h32[16 * HID];    // 8KB : output-precision h
    __shared__ float    cst[16 * HID];    // 8KB : cell state
    __shared__ int      lidx[16 * MAX_LEN];
    __shared__ int      llen[16];

    const int tid  = (int)threadIdx.x;
    const int lane = tid & 31;
    const int w    = tid >> 5;            // wave 0..15
    const int dir  = (int)blockIdx.y;
    const int wg0  = (int)blockIdx.x * 16;

    const float*    Pd = P  + (size_t)dir * N_CHARS * G4;
    const _Float16* Wd = Wt + (size_t)dir * HID * G4;

    if (tid < 256) lidx[tid] = idx[(wg0 + (tid >> 4)) * MAX_LEN + (tid & 15)];
    if (tid < 16)  llen[tid] = lengths[wg0 + tid];
    for (int i = tid; i < 16 * HID; i += 512) {
        h16[i] = (_Float16)0.0f; h32[i] = 0.0f; cst[i] = 0.0f;
    }

    const int nsub = lane & 15;           // tile column / A row
    const int hi   = lane >> 4;           // lane half

    // B fragments: loop-invariant, resident in VGPRs across all 16 steps.
    v16h B[2][4];
#pragma unroll
    for (int ct = 0; ct < 2; ++ct) {
        const int col = (2 * w + ct) * 16 + nsub;
#pragma unroll
        for (int kc = 0; kc < 4; ++kc) {
#pragma unroll
            for (int v = 0; v < 8; ++v) {
                const int k0 = kc * 32 + hi * 16 + 2 * v;
                B[ct][kc][2 * v]     = Wd[(size_t)k0 * G4 + col];
                B[ct][kc][2 * v + 1] = Wd[(size_t)(k0 + 1) * G4 + col];
            }
        }
    }
    __syncthreads();   // lidx/llen/h/c visible

    // prologue: kick off step-0 xproj gather
    async_xproj(Pd, lidx, dir ? (MAX_LEN - 1) : 0, gbuf[0], tid);

    for (int step = 0; step < MAX_LEN; ++step) {
        const int t = dir ? (MAX_LEN - 1 - step) : step;
        float* cur  = gbuf[step & 1];

        // A fragments from LDS h16 (16x32 f16, contiguous k = 16*hi + s, matching B)
        v16h A[4];
        {
            const int m  = nsub;
            const int ko = hi * 16;
#pragma unroll
            for (int kc = 0; kc < 4; ++kc) {
#pragma unroll
                for (int v = 0; v < 8; ++v) {
                    const int off = 2 * v + ko;
                    A[kc][2 * v]     = h16[m * HID + kc * 32 + off];
                    A[kc][2 * v + 1] = h16[m * HID + kc * 32 + off + 1];
                }
            }
        }

        // pipeline: issue next step's gather into the other buffer, then
        // drain only the current step's batch (4 outstanding allowed).
        if (step + 1 < MAX_LEN) {
            async_xproj(Pd, lidx, dir ? (MAX_LEN - 2 - step) : (step + 1),
                        gbuf[(step + 1) & 1], tid);
            WAIT_ASYNCCNT(4);
        } else {
            WAIT_ASYNCCNT(0);
        }
        __syncthreads();   // all waves' xproj slices landed in `cur`

        // C init = xproj tile (bias/input add folded into accumulator)
        v8f C[2];
#pragma unroll
        for (int ct = 0; ct < 2; ++ct) {
            const int col = (2 * w + ct) * 16 + nsub;
#pragma unroll
            for (int v = 0; v < 8; ++v)
                C[ct][v] = cur[(v + 8 * hi) * G4 + col];
        }

        // h @ W_hh^T : 2 col tiles x 4 K-chunks of 32
#pragma unroll
        for (int ct = 0; ct < 2; ++ct) {
#pragma unroll
            for (int kc = 0; kc < 4; ++kc) {
                C[ct] = __builtin_amdgcn_wmma_f32_16x16x32_f16(
                    false, A[kc], false, B[ct][kc], (short)0, C[ct], false, false);
            }
        }
        __syncthreads();   // all waves done reading `cur` before D writes

        // write gate results back to LDS
#pragma unroll
        for (int ct = 0; ct < 2; ++ct) {
            const int col = (2 * w + ct) * 16 + nsub;
#pragma unroll
            for (int v = 0; v < 8; ++v)
                cur[(v + 8 * hi) * G4 + col] = C[ct][v];
        }
        __syncthreads();

        // LSTM cell update with length-mask state freeze (native v_tanh)
        {
            const int j  = tid & 127;
            const int mb = tid >> 7;     // 0..3
#pragma unroll
            for (int r = 0; r < 4; ++r) {
                const int m = r * 4 + mb;
                const float gi = cur[m * G4 + j];
                const float gf = cur[m * G4 + 128 + j];
                const float gg = cur[m * G4 + 256 + j];
                const float go = cur[m * G4 + 384 + j];
                const float ii = fast_sigmoid(gi);
                const float ff = fast_sigmoid(gf);
                const float gv = fast_tanh(gg);
                const float oo = fast_sigmoid(go);
                if (t < llen[m]) {
                    const float cn = ff * cst[m * HID + j] + ii * gv;
                    const float hn = oo * fast_tanh(cn);
                    cst[m * HID + j] = cn;
                    h32[m * HID + j] = hn;
                    h16[m * HID + j] = (_Float16)hn;
                }
            }
        }
        __syncthreads();   // h/c stable; frees `cur` for next step's gather
    }

    // output: [N, 2H], forward cols 0..127, backward 128..255
    {
        const int j  = tid & 127;
        const int mb = tid >> 7;
#pragma unroll
        for (int r = 0; r < 4; ++r) {
            const int m = r * 4 + mb;
            out[(size_t)(wg0 + m) * (2 * HID) + dir * HID + j] = h32[m * HID + j];
        }
    }
}

// ---------------------------------------------------------------------------
extern "C" void kernel_launch(void* const* d_in, const int* in_sizes, int n_in,
                              void* d_out, int out_size, void* d_ws, size_t ws_size,
                              hipStream_t stream) {
    const float* x     = (const float*)d_in[0];
    const int*   len   = (const int*)  d_in[1];
    const float* Wih_f = (const float*)d_in[2];
    const float* Whh_f = (const float*)d_in[3];
    const float* bih_f = (const float*)d_in[4];
    const float* bhh_f = (const float*)d_in[5];
    const float* Wih_b = (const float*)d_in[6];
    const float* Whh_b = (const float*)d_in[7];
    const float* bih_b = (const float*)d_in[8];
    const float* bhh_b = (const float*)d_in[9];

    char* ws = (char*)d_ws;
    const size_t idx_bytes = (size_t)N_WORDS * MAX_LEN * sizeof(int);      // 1 MB
    const size_t P_bytes   = 2ull * N_CHARS * G4 * sizeof(float);          // 512 KB
    int*      idx = (int*)ws;
    float*    P   = (float*)(ws + idx_bytes);
    _Float16* Wt  = (_Float16*)(ws + idx_bytes + P_bytes);                 // 256 KB

    decode_chars_kernel<<<(N_WORDS * MAX_LEN) / 8, 256, 0, stream>>>(x, idx);
    build_tables_kernel<<<256, 512, 0, stream>>>(Wih_f, Whh_f, bih_f, bhh_f,
                                                 Wih_b, Whh_b, bih_b, bhh_b, P, Wt);
    dim3 grid(N_WORDS / 16, 2);
    bilstm_kernel<<<grid, 512, 0, stream>>>(idx, P, Wt, len, (float*)d_out);
}
